// MultiHeadedGraphAttention_R_72138270703638
// MI455X (gfx1250) — matrix operand order repaired and checked by reading, not compiled, not run on hardware
//
#include <hip/hip_runtime.h>

#define B_   4
#define M_   1024
#define D_   512
#define H_   8
#define HD_  64
#define NEMB 300

typedef __attribute__((ext_vector_type(2)))  float   v2f;
typedef __attribute__((ext_vector_type(8)))  float   v8f;
typedef __attribute__((ext_vector_type(8)))  __bf16  v8bf;
typedef __attribute__((ext_vector_type(16))) __bf16  v16bf;

// Build a 16-element bf16 vector (WMMA operand half) from two 16B chunks.
__device__ __forceinline__ v16bf ld_a16(const __bf16* p0, const __bf16* p1) {
  union { v16bf v; v8bf h[2]; } u;
  u.h[0] = *(const v8bf*)p0;
  u.h[1] = *(const v8bf*)p1;
  return u.v;
}

// ---------------------------------------------------------------------------
// pos0[j][d] = sum_i pe[i][j][d]   (pe: [M, M, 64] f32, 256MB stream =
// the dominant HBM term; prefetch the strided column walk)
// ---------------------------------------------------------------------------
__global__ void pos0_kernel(const float* __restrict__ pe, float* __restrict__ pos0) {
  __shared__ float part[256];
  int j = blockIdx.x;
  int t = threadIdx.x;
  int d = t & 63;
  int ic = t >> 6;                       // 0..3, each sums 256 i's
  const float* base = pe + (size_t)j * HD_ + d;
  float acc = 0.f;
  for (int i = ic * 256; i < ic * 256 + 256; ++i) {
    __builtin_prefetch(base + (size_t)(i + 8) * (M_ * HD_), 0, 1);
    acc += base[(size_t)i * (M_ * HD_)];
  }
  part[t] = acc;
  __syncthreads();
  if (t < 64)
    pos0[(size_t)j * HD_ + d] = part[t] + part[t + 64] + part[t + 128] + part[t + 192];
}

// ---------------------------------------------------------------------------
// sq[row] = sum_d v[row][d]^2   (row = b*M + m), one wave per row
// ---------------------------------------------------------------------------
__global__ void sq_kernel(const float* __restrict__ v, float* __restrict__ sq) {
  int w = threadIdx.x >> 5;
  int ln = threadIdx.x & 31;
  int row = blockIdx.x * 8 + w;
  const float* p = v + (size_t)row * D_;
  float acc = 0.f;
  for (int k = ln; k < D_; k += 32) { float x = p[k]; acc += x * x; }
#pragma unroll
  for (int off = 16; off >= 1; off >>= 1) acc += __shfl_xor(acc, off, 32);
  if (ln == 0) sq[row] = acc;
}

// ---------------------------------------------------------------------------
// Gram matrix (full-precision f32 WMMA 16x16x4) -> d2 -> clipped int index u16
// Block: 256 thr (8 waves). 64x64 output tile per block, 2 16x16 tiles/wave.
// ---------------------------------------------------------------------------
__global__ void gram_idx_kernel(const float* __restrict__ v,
                                const float* __restrict__ sq,
                                unsigned short* __restrict__ idxm) {
  __shared__ float vi[64][68];           // row stride 272B (16B aligned)
  __shared__ float vj[64][68];
  int b  = blockIdx.z;
  int i0 = blockIdx.x * 64;
  int j0 = blockIdx.y * 64;
  int tid = threadIdx.x;
  int ln = tid & 31, w = tid >> 5;
  int cn = ln & 15, hi = ln >> 4;
  int tr = w >> 1;                       // tile row 0..3
  int tc0 = (w & 1) * 2;                 // tile cols {0,1} or {2,3}

  const float* vb = v + (size_t)b * M_ * D_;
  v8f acc0 = {}; v8f acc1 = {};
  int ka = hi ? 2 : 0;

  for (int kc = 0; kc < D_ / 64; ++kc) {
    __syncthreads();
    int c4 = (tid & 15) * 4;
    int r0 = tid >> 4;
#pragma unroll
    for (int rr = 0; rr < 4; ++rr) {
      int r = r0 + rr * 16;
      float4 a = *(const float4*)(vb + (size_t)(i0 + r) * D_ + kc * 64 + c4);
      vi[r][c4 + 0] = a.x; vi[r][c4 + 1] = a.y; vi[r][c4 + 2] = a.z; vi[r][c4 + 3] = a.w;
      float4 c = *(const float4*)(vb + (size_t)(j0 + r) * D_ + kc * 64 + c4);
      vj[r][c4 + 0] = c.x; vj[r][c4 + 1] = c.y; vj[r][c4 + 2] = c.z; vj[r][c4 + 3] = c.w;
    }
    __syncthreads();
#pragma unroll
    for (int kk = 0; kk < 16; ++kk) {
      v2f a  = *(const v2f*)&vi[tr * 16 + cn][kk * 4 + ka];
      v2f b0 = *(const v2f*)&vj[tc0 * 16 + cn][kk * 4 + ka];
      v2f b1 = *(const v2f*)&vj[(tc0 + 1) * 16 + cn][kk * 4 + ka];
      acc0 = __builtin_amdgcn_wmma_f32_16x16x4_f32(false, a, false, b0, (short)0, acc0, false, false);
      acc1 = __builtin_amdgcn_wmma_f32_16x16x4_f32(false, a, false, b1, (short)0, acc1, false, false);
    }
  }

  float sqj0 = sq[b * M_ + j0 + tc0 * 16 + cn];
  float sqj1 = sq[b * M_ + j0 + (tc0 + 1) * 16 + cn];
#pragma unroll
  for (int r = 0; r < 8; ++r) {
    int irow = i0 + tr * 16 + r + 8 * hi;
    float sqi = sq[b * M_ + irow];
    float d0 = sqi + sqj0 - 2.f * acc0[r];
    float d1 = sqi + sqj1 - 2.f * acc1[r];
    int e0 = (int)d0; e0 = e0 < 0 ? 0 : (e0 > NEMB - 1 ? NEMB - 1 : e0);
    int e1 = (int)d1; e1 = e1 < 0 ? 0 : (e1 > NEMB - 1 ? NEMB - 1 : e1);
    size_t ro = ((size_t)b * M_ + irow) * M_ + j0;
    idxm[ro + tc0 * 16 + cn]       = (unsigned short)e0;
    idxm[ro + (tc0 + 1) * 16 + cn] = (unsigned short)e1;
  }
}

// ---------------------------------------------------------------------------
// Y = X @ W^T + bias, bf16 WMMA 16x16x32 (f32 accumulate).
// mode 0: khp = kh + pos0  -> bf16 [B,H,M,64]
// mode 1: vhp = vh + pos0  -> bf16 [B,H,M,64]
// mode 2: qh * 0.125       -> bf16 [B,H,M,64]
// mode 3: plain            -> f32 row-major [4096,512] (output projection)
// ---------------------------------------------------------------------------
__global__ void gemm512_kernel(const float* __restrict__ X,
                               const float* __restrict__ W,
                               const float* __restrict__ bias,
                               const float* __restrict__ pos0,
                               void* __restrict__ out,
                               int mode) {
  __shared__ __bf16 xs[64][72];          // row stride 144B (16B aligned)
  __shared__ __bf16 ws_[64][72];
  int t0 = blockIdx.x * 64;              // token tile
  int o0 = blockIdx.y * 64;              // out-feature tile
  int tid = threadIdx.x;
  int ln = tid & 31, w = tid >> 5;
  int cn = ln & 15, hi = ln >> 4;
  int tr = w >> 1, tc0 = (w & 1) * 2;
  int kbA = hi ? 16 : 0;                 // A-layout K base
  int kbB = hi ? 16 : 0;                 // B-layout K base
  v8f acc0 = {}; v8f acc1 = {};

  for (int kc = 0; kc < D_ / 64; ++kc) {
    __syncthreads();
    int c4 = (tid & 15) * 4;
    int r0 = tid >> 4;
#pragma unroll
    for (int rr = 0; rr < 4; ++rr) {
      int r = r0 + rr * 16;
      float4 a = *(const float4*)(X + (size_t)(t0 + r) * D_ + kc * 64 + c4);
      xs[r][c4 + 0] = (__bf16)a.x; xs[r][c4 + 1] = (__bf16)a.y;
      xs[r][c4 + 2] = (__bf16)a.z; xs[r][c4 + 3] = (__bf16)a.w;
      float4 b = *(const float4*)(W + (size_t)(o0 + r) * D_ + kc * 64 + c4);
      ws_[r][c4 + 0] = (__bf16)b.x; ws_[r][c4 + 1] = (__bf16)b.y;
      ws_[r][c4 + 2] = (__bf16)b.z; ws_[r][c4 + 3] = (__bf16)b.w;
    }
    __syncthreads();
    int arow = tr * 16 + cn;
    v16bf a0 = ld_a16(&xs[arow][kbA],      &xs[arow][kbA + 8]);
    v16bf a1 = ld_a16(&xs[arow][32 + kbA], &xs[arow][32 + kbA + 8]);
    int bc0 = tc0 * 16 + cn, bc1 = bc0 + 16;
    v16bf b00 = ld_a16(&ws_[bc0][kbB],      &ws_[bc0][kbB + 8]);
    v16bf b01 = ld_a16(&ws_[bc0][32 + kbB], &ws_[bc0][32 + kbB + 8]);
    v16bf b10 = ld_a16(&ws_[bc1][kbB],      &ws_[bc1][kbB + 8]);
    v16bf b11 = ld_a16(&ws_[bc1][32 + kbB], &ws_[bc1][32 + kbB + 8]);
    acc0 = __builtin_amdgcn_wmma_f32_16x16x32_bf16(false, a0, false, b00, (short)0, acc0, false, false);
    acc0 = __builtin_amdgcn_wmma_f32_16x16x32_bf16(false, a1, false, b01, (short)0, acc0, false, false);
    acc1 = __builtin_amdgcn_wmma_f32_16x16x32_bf16(false, a0, false, b10, (short)0, acc1, false, false);
    acc1 = __builtin_amdgcn_wmma_f32_16x16x32_bf16(false, a1, false, b11, (short)0, acc1, false, false);
  }

  int oc0 = o0 + tc0 * 16 + cn, oc1 = oc0 + 16;
  float bias0 = bias[oc0], bias1 = bias[oc1];
#pragma unroll
  for (int r = 0; r < 8; ++r) {
    int tok = t0 + tr * 16 + r + 8 * hi;
    int bb = tok >> 10, mm = tok & 1023;
    float y0 = acc0[r] + bias0;
    float y1 = acc1[r] + bias1;
    if (mode == 0 || mode == 1) {
      y0 += pos0[(size_t)mm * HD_ + (oc0 & 63)];
      y1 += pos0[(size_t)mm * HD_ + (oc1 & 63)];
    } else if (mode == 2) {
      y0 *= 0.125f; y1 *= 0.125f;          // 1/sqrt(64)
    }
    if (mode == 3) {
      ((float*)out)[(size_t)tok * 512 + oc0] = y0;
      ((float*)out)[(size_t)tok * 512 + oc1] = y1;
    } else {
      __bf16* ob = (__bf16*)out;
      ob[(((size_t)bb * H_ + (oc0 >> 6)) * M_ + mm) * HD_ + (oc0 & 63)] = (__bf16)y0;
      ob[(((size_t)bb * H_ + (oc1 >> 6)) * M_ + mm) * HD_ + (oc1 & 63)] = (__bf16)y1;
    }
  }
}

// ---------------------------------------------------------------------------
// Fused flash attention on bf16 q/k/v: S = qh @ khp^T + r_emb[idx] (+mask),
// online softmax, O = P @ vhp.  Block: 128 thr (4 waves), wave owns 16 query
// rows x all hd.  Grid: (M/64, H, B).  Streams j in 64-wide chunks; staging
// is b128 loads; K copied, V transposed (inputs already bf16).
// ---------------------------------------------------------------------------
__global__ void attn_kernel(const __bf16* __restrict__ qh,
                            const __bf16* __restrict__ khp,
                            const __bf16* __restrict__ vhp,
                            const unsigned short* __restrict__ idxm,
                            const float* __restrict__ r_emb,
                            const unsigned char* __restrict__ mask,
                            float* __restrict__ ctx) {
  __shared__ __bf16 ks[64][72];          // khp chunk, row-major [j][hd]
  __shared__ __bf16 vts[64][72];         // vhp chunk, transposed [hd][j]
  __shared__ __bf16 ps[4][16][72];       // per-wave probs [row][j]
  __shared__ float remb_s[NEMB];
  __shared__ unsigned char mask_s[64];

  int i0 = blockIdx.x * 64;
  int h  = blockIdx.y;
  int b  = blockIdx.z;
  int tid = threadIdx.x;
  int ln = tid & 31, w = tid >> 5;
  int cn = ln & 15, hi = ln >> 4;
  int kbA = hi ? 16 : 0;
  int kbB = hi ? 16 : 0;

  for (int e = tid; e < NEMB; e += 128) remb_s[e] = r_emb[(size_t)e * H_ + h];

  const size_t headoff = ((size_t)b * H_ + h) * (size_t)M_ * HD_;

  // Q already bf16 in WMMA-friendly row-major: load A operands directly.
  const __bf16* qp = qh + headoff + (size_t)(i0 + w * 16 + cn) * HD_;
  v16bf aq0 = ld_a16(qp + kbA,      qp + kbA + 8);
  v16bf aq1 = ld_a16(qp + 32 + kbA, qp + 32 + kbA + 8);

  float mst[8], lst[8];
  v8f zero = {};
  v8f oacc[4];
#pragma unroll
  for (int r = 0; r < 8; ++r) { mst[r] = -3.0e38f; lst[r] = 0.f; }
#pragma unroll
  for (int t = 0; t < 4; ++t) oacc[t] = zero;

  for (int jc = 0; jc < M_ / 64; ++jc) {
    int j0 = jc * 64;
    __syncthreads();
    {
      // 64 rows x 64 bf16 = 512 16B-chunks per buffer; 4 chunks per thread.
      int c8  = tid & 7;                  // 16B chunk within row
      int jr0 = tid >> 3;                 // 0..15
      const __bf16* kpg = khp + headoff + (size_t)j0 * HD_;
      const __bf16* vpg = vhp + headoff + (size_t)j0 * HD_;
#pragma unroll
      for (int rr = 0; rr < 4; ++rr) {
        int jr = jr0 + rr * 16;
        *(v8bf*)&ks[jr][c8 * 8] = *(const v8bf*)(kpg + (size_t)jr * HD_ + c8 * 8);
        v8bf vv = *(const v8bf*)(vpg + (size_t)jr * HD_ + c8 * 8);
#pragma unroll
        for (int e = 0; e < 8; ++e) vts[c8 * 8 + e][jr] = vv[e];
      }
      if (tid < 64) mask_s[tid] = mask[(size_t)b * M_ + j0 + tid];
    }
    __syncthreads();

    // S tiles
    v8f s[4];
#pragma unroll
    for (int t = 0; t < 4; ++t) {
      int col = t * 16 + cn;
      v16bf b0 = ld_a16(&ks[col][kbB],      &ks[col][kbB + 8]);
      v16bf b1 = ld_a16(&ks[col][32 + kbB], &ks[col][32 + kbB + 8]);
      v8f a = zero;
      a = __builtin_amdgcn_wmma_f32_16x16x32_bf16(false, aq0, false, b0, (short)0, a, false, false);
      a = __builtin_amdgcn_wmma_f32_16x16x32_bf16(false, aq1, false, b1, (short)0, a, false, false);
      s[t] = a;
    }

    // distance-embedding bias + key mask
#pragma unroll
    for (int t = 0; t < 4; ++t) {
      int jcol = j0 + t * 16 + cn;
      bool mk = mask_s[t * 16 + cn] != 0;
#pragma unroll
      for (int r = 0; r < 8; ++r) {
        int ir = i0 + w * 16 + r + 8 * hi;
        unsigned short e = idxm[((size_t)b * M_ + ir) * M_ + jcol];
        float val = s[t][r] + remb_s[e];
        s[t][r] = mk ? val : -3.0e38f;
      }
    }

    // online softmax (row reductions stay inside 16-lane groups)
    float scl[8];
#pragma unroll
    for (int r = 0; r < 8; ++r) {
      float mx = fmaxf(fmaxf(s[0][r], s[1][r]), fmaxf(s[2][r], s[3][r]));
#pragma unroll
      for (int off = 8; off >= 1; off >>= 1) mx = fmaxf(mx, __shfl_xor(mx, off, 32));
      float mnew = fmaxf(mst[r], mx);
      float sc = __expf(mst[r] - mnew);
      float rs = 0.f;
#pragma unroll
      for (int t = 0; t < 4; ++t) {
        float p = __expf(s[t][r] - mnew);
        s[t][r] = p;
        rs += p;
      }
#pragma unroll
      for (int off = 8; off >= 1; off >>= 1) rs += __shfl_xor(rs, off, 32);
      lst[r] = lst[r] * sc + rs;
      mst[r] = mnew;
      scl[r] = sc;
    }
#pragma unroll
    for (int t = 0; t < 4; ++t)
#pragma unroll
      for (int r = 0; r < 8; ++r) oacc[t][r] *= scl[r];

    // transpose P (C layout -> A layout) through per-wave LDS
#pragma unroll
    for (int t = 0; t < 4; ++t)
#pragma unroll
      for (int r = 0; r < 8; ++r)
        ps[w][r + 8 * hi][t * 16 + cn] = (__bf16)s[t][r];

    v16bf ap0 = ld_a16(&ps[w][cn][kbA],      &ps[w][cn][kbA + 8]);
    v16bf ap1 = ld_a16(&ps[w][cn][32 + kbA], &ps[w][cn][32 + kbA + 8]);
#pragma unroll
    for (int t = 0; t < 4; ++t) {
      int n = t * 16 + cn;
      v16bf b0 = ld_a16(&vts[n][kbB],      &vts[n][kbB + 8]);
      v16bf b1 = ld_a16(&vts[n][32 + kbB], &vts[n][32 + kbB + 8]);
      v8f o = oacc[t];
      o = __builtin_amdgcn_wmma_f32_16x16x32_bf16(false, ap0, false, b0, (short)0, o, false, false);
      o = __builtin_amdgcn_wmma_f32_16x16x32_bf16(false, ap1, false, b1, (short)0, o, false, false);
      oacc[t] = o;
    }
  }

  // normalize and write context, token-major [B, M, H*64] f32
#pragma unroll
  for (int t = 0; t < 4; ++t) {
    int n = t * 16 + cn;
#pragma unroll
    for (int r = 0; r < 8; ++r) {
      int ir = i0 + w * 16 + r + 8 * hi;
      ctx[((size_t)b * M_ + ir) * 512 + h * HD_ + n] = oacc[t][r] / lst[r];
    }
  }
}

extern "C" void kernel_launch(void* const* d_in, const int* in_sizes, int n_in,
                              void* d_out, int out_size, void* d_ws, size_t ws_size,
                              hipStream_t stream) {
  (void)in_sizes; (void)n_in; (void)out_size; (void)ws_size;
  const float* k    = (const float*)d_in[0];
  const float* v    = (const float*)d_in[1];
  const float* q    = (const float*)d_in[2];
  const unsigned char* mask = (const unsigned char*)d_in[3];
  const float* pe   = (const float*)d_in[4];
  const float* Wk   = (const float*)d_in[5];
  const float* bk   = (const float*)d_in[6];
  const float* Wv   = (const float*)d_in[7];
  const float* bv   = (const float*)d_in[8];
  const float* Wq   = (const float*)d_in[9];
  const float* bq   = (const float*)d_in[10];
  const float* Wo   = (const float*)d_in[11];
  const float* bo   = (const float*)d_in[12];
  const float* remb = (const float*)d_in[13];
  float* out = (float*)d_out;

  char* ws = (char*)d_ws;
  size_t off = 0;
  auto alloc = [&](size_t bytes) -> void* {
    void* p = ws + off;
    off = (off + bytes + 255) & ~(size_t)255;
    return p;
  };
  float*          pos0 = (float*)alloc((size_t)M_ * HD_ * 4);
  float*          sqv  = (float*)alloc((size_t)B_ * M_ * 4);
  unsigned short* idxm = (unsigned short*)alloc((size_t)B_ * M_ * M_ * 2);
  __bf16*         khp  = (__bf16*)alloc((size_t)B_ * H_ * M_ * HD_ * 2);
  __bf16*         vhp  = (__bf16*)alloc((size_t)B_ * H_ * M_ * HD_ * 2);
  __bf16*         qhp  = (__bf16*)alloc((size_t)B_ * H_ * M_ * HD_ * 2);
  float*          ctx  = (float*)alloc((size_t)B_ * M_ * 512 * 4);

  pos0_kernel<<<dim3(M_), dim3(256), 0, stream>>>(pe, pos0);
  sq_kernel<<<dim3(B_ * M_ / 8), dim3(256), 0, stream>>>(v, sqv);
  gram_idx_kernel<<<dim3(16, 16, B_), dim3(256), 0, stream>>>(v, sqv, idxm);
  gemm512_kernel<<<dim3(64, 8), dim3(256), 0, stream>>>(k, Wk, bk, pos0, khp, 0);
  gemm512_kernel<<<dim3(64, 8), dim3(256), 0, stream>>>(v, Wv, bv, pos0, vhp, 1);
  gemm512_kernel<<<dim3(64, 8), dim3(256), 0, stream>>>(q, Wq, bq, pos0, qhp, 2);
  attn_kernel<<<dim3(M_ / 64, H_, B_), dim3(128), 0, stream>>>(qhp, khp, vhp, idxm, remb, mask, ctx);
  gemm512_kernel<<<dim3(64, 8), dim3(256), 0, stream>>>(ctx, Wo, bo, pos0, out, 3);
}
